// MixedFFN_5772436046040
// MI455X (gfx1250) — matrix operand order repaired and checked by reading, not compiled, not run on hardware
//
#include <hip/hip_runtime.h>
#include <hip/hip_bf16.h>
#include <cstdint>

// MixedFFN for MI455X (gfx1250, wave32).
// B=8, T=2048, D=1024, F=4096, LNS=32 -> 16128 shared tokens + 32 per-token FFNs.
// Two-pass bf16-WMMA FFN (v_wmma_f32_16x16x32_bf16, fp32 accumulate);
// hidden H staged as bf16 in d_ws:
//   H_shared [16128][4096] bf16 (132 MB), then H_tok [32][16][4096] bf16 (4 MB).

typedef __bf16 bf16;
typedef __attribute__((ext_vector_type(16))) __bf16 v16bf;
typedef __attribute__((ext_vector_type(8)))  __bf16 bf16x8;
typedef __attribute__((ext_vector_type(8)))  float  v8f;
typedef __attribute__((ext_vector_type(4)))  float  f4;

#define AK 40   // LDS row stride in bf16 (80B: 16B-aligned, conflict-breaking)

// Branch-free exact-GELU: erf via Abramowitz-Stegun 7.1.26 (|err|<1.5e-7).
// One v_rcp + one v_exp + ~10 FMA; low register pressure (no scratch spills).
__device__ __forceinline__ float gelu_exact(float v) {
    float x  = v * 0.70710678118654752f;
    float ax = __builtin_fabsf(x);
    float t  = __builtin_amdgcn_rcpf(__builtin_fmaf(0.3275911f, ax, 1.0f));
    float p  = __builtin_fmaf(t, 1.061405429f, -1.453152027f);
    p = __builtin_fmaf(t, p, 1.421413741f);
    p = __builtin_fmaf(t, p, -0.284496736f);
    p = __builtin_fmaf(t, p, 0.254829592f);
    float e   = __expf(-ax * ax);
    float erf = __builtin_fmaf(-p * t, e, 1.0f);
    erf = __builtin_copysignf(erf, x);
    return 0.5f * v * (1.0f + erf);
}

__device__ __forceinline__ v8f wmma_bf16(v16bf a, v16bf b, v8f c) {
    return __builtin_amdgcn_wmma_f32_16x16x32_bf16(false, a, false, b, (short)0, c, false, false);
}

// Pack 4 converted bf16 and store as a single 8-byte LDS write (ds_store_b64).
__device__ __forceinline__ void store4bf16(bf16* p, f4 v) {
    union { bf16 h[4]; unsigned long long u; } u;
    u.h[0] = (bf16)v.x; u.h[1] = (bf16)v.y; u.h[2] = (bf16)v.z; u.h[3] = (bf16)v.w;
    *(unsigned long long*)p = u.u;
}

// A fragment: 16x32 bf16, rows r0..r0+15, LDS row-major [row][k].
// lane<16: row=lane, K={0..7,16..23}; lane>=16: row=lane-16, K={8..15,24..31}
__device__ __forceinline__ v16bf load_frag_A(const bf16* As, int lane, int r0) {
    int lrow = lane & 15, half = lane >> 4;
    const bf16* p = As + (r0 + lrow) * AK + half * 8;
    union { v16bf v; bf16x8 h[2]; } u;
    u.h[0] = *(const bf16x8*)p;
    u.h[1] = *(const bf16x8*)(p + 16);
    return u.v;
}

// B fragment: 32x16 bf16 (KxN), LDS stored transposed [n][k].
// lane<16: n=lane, K={0..7,16..23} ; lane>=16: n=lane-16, K={8..15,24..31}
__device__ __forceinline__ v16bf load_frag_B(const bf16* Bs, int lane, int c0) {
    int n = lane & 15, kg = (lane >> 4) * 8;
    const bf16* p = Bs + (c0 + n) * AK + kg;
    union { v16bf v; bf16x8 h[2]; } u;
    u.h[0] = *(const bf16x8*)p;
    u.h[1] = *(const bf16x8*)(p + 16);
    return u.v;
}

// ---------------- Pass 1 (shared): H = gelu(X @ W1S), M=16128 K=1024 N=4096 ----
__global__ __launch_bounds__(256) void k_gemm1_shared(
    const float* __restrict__ x, const float* __restrict__ W1, bf16* __restrict__ H) {
    __shared__ bf16 As[128 * AK];
    __shared__ bf16 Bs[128 * AK];
    const int n0 = blockIdx.x * 128, m0 = blockIdx.y * 128;
    const int tid = threadIdx.x, lane = tid & 31, wave = tid >> 5;
    const int wr = wave >> 1, wc = wave & 1;      // wave -> 32-row x 64-col patch
    v8f acc[2][4] = {};

    // Hoisted staging addresses (advance by constant stride per k-step).
    const float* gpa[4]; bf16* lpa[4];
    const float* gpb[4]; bf16* lpb[4];
    #pragma unroll
    for (int i = 0; i < 4; ++i) {
        int c = tid + i * 256;
        int rowA = c >> 3, col4 = (c & 7) * 4;
        int m = m0 + rowA, b = m / 2016, srow = m - b * 2016;
        gpa[i] = x + ((size_t)(b * 2048 + srow)) * 1024 + col4;
        lpa[i] = As + rowA * AK + col4;
        int k = c >> 5, n4 = (c & 31) * 4;
        gpb[i] = W1 + (size_t)k * 4096 + n0 + n4;
        lpb[i] = Bs + n4 * AK + k;
    }

    for (int k0 = 0; k0 < 1024; k0 += 32) {
        __syncthreads();
        #pragma unroll
        for (int i = 0; i < 4; ++i) {              // A: 128x32 fp32 -> bf16
            f4 v = *(const f4*)gpa[i]; gpa[i] += 32;
            store4bf16(lpa[i], v);
        }
        #pragma unroll
        for (int i = 0; i < 4; ++i) {              // B: 32(k)x128(n) -> Bs[n][k]
            f4 v = *(const f4*)gpb[i]; gpb[i] += (size_t)32 * 4096;
            lpb[i][0 * AK] = (bf16)v.x;
            lpb[i][1 * AK] = (bf16)v.y;
            lpb[i][2 * AK] = (bf16)v.z;
            lpb[i][3 * AK] = (bf16)v.w;
        }
        __syncthreads();
        v16bf a0 = load_frag_A(As, lane, wr * 32);
        v16bf a1 = load_frag_A(As, lane, wr * 32 + 16);
        #pragma unroll
        for (int j = 0; j < 4; ++j) {
            v16bf bf = load_frag_B(Bs, lane, wc * 64 + j * 16);
            acc[0][j] = wmma_bf16(a0, bf, acc[0][j]);
            acc[1][j] = wmma_bf16(a1, bf, acc[1][j]);
        }
    }
    // Epilogue: gelu -> bf16 H[m][f]
    int colb = lane & 15, rowb = (lane >> 4) * 8;
    #pragma unroll
    for (int r = 0; r < 2; ++r)
        #pragma unroll
        for (int j = 0; j < 4; ++j)
            #pragma unroll
            for (int i = 0; i < 8; ++i) {
                int gr = m0 + wr * 32 + r * 16 + rowb + i;
                int gc = n0 + wc * 64 + j * 16 + colb;
                H[(size_t)gr * 4096 + gc] = (bf16)gelu_exact(acc[r][j][i]);
            }
}

// ---------------- Pass 2 (shared): Y = H @ W2S, M=16128 K=4096 N=1024 ----------
__global__ __launch_bounds__(256) void k_gemm2_shared(
    const bf16* __restrict__ H, const float* __restrict__ W2, float* __restrict__ out) {
    __shared__ bf16 As[128 * AK];
    __shared__ bf16 Bs[128 * AK];
    const int n0 = blockIdx.x * 128, m0 = blockIdx.y * 128;
    const int tid = threadIdx.x, lane = tid & 31, wave = tid >> 5;
    const int wr = wave >> 1, wc = wave & 1;
    v8f acc[2][4] = {};

    unsigned long long gpa[2]; unsigned int loffA[2];
    const float* gpb[4]; bf16* lpb[4];
    #pragma unroll
    for (int i = 0; i < 2; ++i) {                  // A tile: 512 x 16B async chunks
        int c = tid + i * 256;
        int row = c >> 2, col = (c & 3) * 8;
        gpa[i]   = (unsigned long long)(uintptr_t)(H + (size_t)(m0 + row) * 4096 + col);
        loffA[i] = (unsigned int)(uintptr_t)(As + row * AK + col);
    }
    #pragma unroll
    for (int i = 0; i < 4; ++i) {
        int c = tid + i * 256;
        int k = c >> 5, n4 = (c & 31) * 4;
        gpb[i] = W2 + (size_t)k * 1024 + n0 + n4;
        lpb[i] = Bs + n4 * AK + k;
    }

    for (int k0 = 0; k0 < 4096; k0 += 32) {
        __syncthreads();
        #pragma unroll
        for (int i = 0; i < 2; ++i) {              // async global->LDS (ASYNCcnt)
            asm volatile("global_load_async_to_lds_b128 %0, %1, off"
                         :: "v"(loffA[i]), "v"(gpa[i]) : "memory");
            gpa[i] += 64;                          // 32 bf16 per k-step
        }
        #pragma unroll
        for (int i = 0; i < 4; ++i) {              // B: fp32 -> bf16 transposed
            f4 v = *(const f4*)gpb[i]; gpb[i] += (size_t)32 * 1024;
            lpb[i][0 * AK] = (bf16)v.x;
            lpb[i][1 * AK] = (bf16)v.y;
            lpb[i][2 * AK] = (bf16)v.z;
            lpb[i][3 * AK] = (bf16)v.w;
        }
        asm volatile("s_wait_asynccnt 0" ::: "memory");
        __syncthreads();
        v16bf a0 = load_frag_A(As, lane, wr * 32);
        v16bf a1 = load_frag_A(As, lane, wr * 32 + 16);
        #pragma unroll
        for (int j = 0; j < 4; ++j) {
            v16bf bf = load_frag_B(Bs, lane, wc * 64 + j * 16);
            acc[0][j] = wmma_bf16(a0, bf, acc[0][j]);
            acc[1][j] = wmma_bf16(a1, bf, acc[1][j]);
        }
    }
    int colb = lane & 15, rowb = (lane >> 4) * 8;
    #pragma unroll
    for (int r = 0; r < 2; ++r)
        #pragma unroll
        for (int j = 0; j < 4; ++j)
            #pragma unroll
            for (int i = 0; i < 8; ++i) {
                int m = m0 + wr * 32 + r * 16 + rowb + i;
                int b = m / 2016, srow = m - b * 2016;
                int gc = n0 + wc * 64 + j * 16 + colb;
                out[((size_t)(b * 2048 + srow)) * 1024 + gc] = acc[r][j][i];
            }
}

// ------------- Pass 1 (token): Ht[tk] = gelu(Xt[tk] @ W1NS[tk]), M=16(pad) -----
__global__ __launch_bounds__(256) void k_gemm1_tok(
    const float* __restrict__ x, const float* __restrict__ W1NS, bf16* __restrict__ Ht) {
    __shared__ bf16 As[16 * AK];
    __shared__ bf16 Bs[128 * AK];
    const int n0 = blockIdx.x * 128, tk = blockIdx.y;
    const int tid = threadIdx.x, lane = tid & 31, wave = tid >> 5;
    v8f acc = {};

    const float* gpa = nullptr; bf16* lpa = nullptr;
    bool arow = false;
    if (tid < 128) {                               // A: 16x32, rows 8..15 zero pad
        int row = tid >> 3, col4 = (tid & 7) * 4;
        arow = (row < 8);
        gpa = x + ((size_t)(row * 2048 + 2016 + tk)) * 1024 + col4;
        lpa = As + row * AK + col4;
    }
    const float* gpb[4]; bf16* lpb[4];
    const float* W1 = W1NS + (size_t)tk * 1024 * 4096;
    #pragma unroll
    for (int i = 0; i < 4; ++i) {
        int c = tid + i * 256;
        int k = c >> 5, n4 = (c & 31) * 4;
        gpb[i] = W1 + (size_t)k * 4096 + n0 + n4;
        lpb[i] = Bs + n4 * AK + k;
    }

    for (int k0 = 0; k0 < 1024; k0 += 32) {
        __syncthreads();
        if (tid < 128) {
            f4 v = {};
            if (arow) v = *(const f4*)gpa;
            gpa += 32;
            store4bf16(lpa, v);
        }
        #pragma unroll
        for (int i = 0; i < 4; ++i) {              // per-token weights (streamed)
            f4 v = *(const f4*)gpb[i];
            gpb[i] += (size_t)32 * 4096;
            __builtin_prefetch(gpb[i], 0, 0);      // global_prefetch_b8: next tile
            lpb[i][0 * AK] = (bf16)v.x;
            lpb[i][1 * AK] = (bf16)v.y;
            lpb[i][2 * AK] = (bf16)v.z;
            lpb[i][3 * AK] = (bf16)v.w;
        }
        __syncthreads();
        v16bf a = load_frag_A(As, lane, 0);
        v16bf b = load_frag_B(Bs, lane, wave * 16);
        acc = wmma_bf16(a, b, acc);
    }
    int colb = lane & 15, rowb = (lane >> 4) * 8;
    #pragma unroll
    for (int i = 0; i < 8; ++i) {
        int r = rowb + i, gc = n0 + wave * 16 + colb;
        Ht[((size_t)tk * 16 + r) * 4096 + gc] = (bf16)gelu_exact(acc[i]);
    }
}

// ------------- Pass 2 (token): Yt[tk] = Ht[tk] @ W2NS[tk], K=4096 --------------
__global__ __launch_bounds__(256) void k_gemm2_tok(
    const bf16* __restrict__ Ht, const float* __restrict__ W2NS, float* __restrict__ out) {
    __shared__ bf16 As[16 * AK];
    __shared__ bf16 Bs[128 * AK];
    const int n0 = blockIdx.x * 128, tk = blockIdx.y;
    const int tid = threadIdx.x, lane = tid & 31, wave = tid >> 5;
    v8f acc = {};

    const bf16* gpa = nullptr; bf16* lpa = nullptr;
    if (tid < 64) {                                // A: 16x32 bf16, contiguous
        int row = tid >> 2, col = (tid & 3) * 8;
        gpa = Ht + ((size_t)tk * 16 + row) * 4096 + col;
        lpa = As + row * AK + col;
    }
    const float* gpb[4]; bf16* lpb[4];
    const float* W2 = W2NS + (size_t)tk * 4096 * 1024;
    #pragma unroll
    for (int i = 0; i < 4; ++i) {
        int c = tid + i * 256;
        int k = c >> 5, n4 = (c & 31) * 4;
        gpb[i] = W2 + (size_t)k * 1024 + n0 + n4;
        lpb[i] = Bs + n4 * AK + k;
    }

    for (int k0 = 0; k0 < 4096; k0 += 32) {
        __syncthreads();
        if (tid < 64) {
            *(bf16x8*)lpa = *(const bf16x8*)gpa;
            gpa += 32;
        }
        #pragma unroll
        for (int i = 0; i < 4; ++i) {              // per-token weights (streamed)
            f4 v = *(const f4*)gpb[i];
            gpb[i] += (size_t)32 * 1024;
            __builtin_prefetch(gpb[i], 0, 0);      // global_prefetch_b8: next tile
            lpb[i][0 * AK] = (bf16)v.x;
            lpb[i][1 * AK] = (bf16)v.y;
            lpb[i][2 * AK] = (bf16)v.z;
            lpb[i][3 * AK] = (bf16)v.w;
        }
        __syncthreads();
        v16bf a = load_frag_A(As, lane, 0);
        v16bf b = load_frag_B(Bs, lane, wave * 16);
        acc = wmma_bf16(a, b, acc);
    }
    int colb = lane & 15, rowb = (lane >> 4) * 8;
    #pragma unroll
    for (int i = 0; i < 8; ++i) {
        int r = rowb + i;
        if (r < 8) {                               // only real batch rows
            out[((size_t)(r * 2048 + 2016 + tk)) * 1024 + n0 + wave * 16 + colb] = acc[i];
        }
    }
}

extern "C" void kernel_launch(void* const* d_in, const int* in_sizes, int n_in,
                              void* d_out, int out_size, void* d_ws, size_t ws_size,
                              hipStream_t stream) {
    const float* x    = (const float*)d_in[0];   // (8,2048,1024)
    const float* W1S  = (const float*)d_in[1];   // (1024,4096)
    const float* W2S  = (const float*)d_in[2];   // (4096,1024)
    const float* W1NS = (const float*)d_in[3];   // (32,1024,4096)
    const float* W2NS = (const float*)d_in[4];   // (32,4096,1024)
    float* out = (float*)d_out;                  // (8,2048,1024) fp32

    bf16* H  = (bf16*)d_ws;                      // [16128][4096]
    bf16* Ht = H + (size_t)16128 * 4096;         // [32][16][4096]

    dim3 blk(256);
    k_gemm1_shared<<<dim3(32, 126), blk, 0, stream>>>(x, W1S, H);
    k_gemm1_tok   <<<dim3(32, 32),  blk, 0, stream>>>(x, W1NS, Ht);
    k_gemm2_shared<<<dim3(8, 126),  blk, 0, stream>>>(H, W2S, out);
    k_gemm2_tok   <<<dim3(8, 32),   blk, 0, stream>>>(Ht, W2NS, out);
}
